// Generator_738734374998
// MI455X (gfx1250) — compile-verified, matrix operand order checked by reading, not run on hardware
//
#include <hip/hip_runtime.h>
#include <hip/hip_bf16.h>
#include <math.h>

// Problem constants (match reference)
#define Vv   18000
#define Hh   400
#define HPAD 416                // 13 * 32, zero-padded K for bf16 WMMA
#define HCH  (HPAD / 8)         // 52 x 8-element chunks per row
#define Ss   30
#define Bb   16
#define Ll   400
#define Tt   10
#define NGg  3
#define SBr  (Ss * Bb)          // 480 rows
#define H3   (3 * Hh)
#define NEGINF (-1e30f)

// WMMA vector types (gfx1250, wave32)
typedef __bf16 v16bf __attribute__((ext_vector_type(16)));
typedef __bf16 v8bf  __attribute__((ext_vector_type(8)));   // 16 bytes
typedef float  v8f   __attribute__((ext_vector_type(8)));

// ---------------------------------------------------------------------------
// block-wide reduction helper (256 threads)
__device__ __forceinline__ float block_reduce(float v, float* red, int is_max) {
    int tid = threadIdx.x;
    red[tid] = v;
    __syncthreads();
    for (int s = 128; s > 0; s >>= 1) {
        if (tid < s) red[tid] = is_max ? fmaxf(red[tid], red[tid + s])
                                       : (red[tid] + red[tid + s]);
        __syncthreads();
    }
    float r = red[0];
    __syncthreads();
    return r;
}

// ---------------------------------------------------------------------------
// Kernel 0 (once per launch): emb_weight f32 [V,400] -> bf16 [V,416], pad = 0
__global__ void __launch_bounds__(256) emb_to_bf16_kernel(
    const float* __restrict__ emb, __bf16* __restrict__ embb)
{
    const int idx = blockIdx.x * 256 + threadIdx.x;   // one 8-elem chunk each
    if (idx >= Vv * HCH) return;
    const int n  = idx / HCH;
    const int cb = (idx % HCH) * 8;
    v8bf o;
    if (cb < Hh) {   // full 8-wide chunks only (400 % 8 == 0)
        const float4 f0 = *(const float4*)(emb + (size_t)n * Hh + cb);
        const float4 f1 = *(const float4*)(emb + (size_t)n * Hh + cb + 4);
        o[0] = (__bf16)f0.x; o[1] = (__bf16)f0.y; o[2] = (__bf16)f0.z; o[3] = (__bf16)f0.w;
        o[4] = (__bf16)f1.x; o[5] = (__bf16)f1.y; o[6] = (__bf16)f1.z; o[7] = (__bf16)f1.w;
    } else {
        #pragma unroll
        for (int i = 0; i < 8; ++i) o[i] = (__bf16)0.0f;
    }
    *(v8bf*)(embb + (size_t)n * HPAD + cb) = o;
}

// ---------------------------------------------------------------------------
// Kernel 1: GRU cell + masked dot-attention + pointer switch (+ gate at t==0)
// one block (256 thr) per decoder row sb = s*B + b
__global__ void __launch_bounds__(256) gru_attn_kernel(
    const float* __restrict__ enc_hidden,   // [1,B,H]
    const float* __restrict__ enc_out,      // [B,L,H]
    const int*   __restrict__ enc_len,      // [B]
    const int*   __restrict__ target,       // [S,B,T]
    const int*   __restrict__ domain_idx,   // [S]
    const int*   __restrict__ slot_idx,     // [S]
    const float* __restrict__ emb,          // [V,H]
    const float* __restrict__ slot_tab,     // [36,H]
    const float* __restrict__ w_ih,         // [3H,H]
    const float* __restrict__ w_hh,         // [3H,H]
    const float* __restrict__ b_ih,         // [3H]
    const float* __restrict__ b_hh,         // [3H]
    const float* __restrict__ w_ratio_w,    // [1,3H]
    const float* __restrict__ w_ratio_b,    // [1]
    const float* __restrict__ w_gate_w,     // [NG,H]
    const float* __restrict__ w_gate_b,     // [NG]
    const float* __restrict__ h_prev,       // [SB,H]  f32
    float*       __restrict__ h_new,        // [SB,H]  f32 (next GRU step)
    __bf16*      __restrict__ h_new_b,      // [SB,HPAD] bf16 (WMMA A operand)
    float*       __restrict__ prob_g,       // [SB,L]
    float*       __restrict__ sw_g,         // [SB]
    float*       __restrict__ gate_out,     // [SB,NG]
    int t)
{
    __shared__ float xs[Hh], hs[Hh], hn[Hh], cx[Hh];
    __shared__ float sc[Ll];
    __shared__ float red[256];

    const int row = blockIdx.x;
    const int tid = threadIdx.x;
    const int b   = row % Bb;        // hidden/enc rows are s*B+b ordered

    // ---- decoder input x
    if (t == 0) {
        const int s  = row / Bb;
        const int d  = domain_idx[s];
        const int sl = slot_idx[s];
        for (int j = tid; j < Hh; j += 256)
            xs[j] = slot_tab[d * Hh + j] + slot_tab[sl * Hh + j];
        for (int j = tid; j < Hh; j += 256)
            hs[j] = enc_hidden[b * Hh + j];
    } else {
        // replicate repo quirk: dec input rows are flattened b*S+s
        const int bb = row / Ss;
        const int ss = row % Ss;
        const int tok = target[(ss * Bb + bb) * Tt + (t - 1)];
        for (int j = tid; j < Hh; j += 256)
            xs[j] = emb[(size_t)tok * Hh + j];
        for (int j = tid; j < Hh; j += 256)
            hs[j] = h_prev[row * Hh + j];
    }
    __syncthreads();

    // ---- GRU cell (torch semantics)
    for (int j = tid; j < Hh; j += 256) {
        float ir = b_ih[j], iz = b_ih[Hh + j], inn = b_ih[2 * Hh + j];
        float hr = b_hh[j], hz = b_hh[Hh + j], hnn = b_hh[2 * Hh + j];
        const float* wir = w_ih + (size_t)j * Hh;
        const float* wiz = w_ih + (size_t)(Hh + j) * Hh;
        const float* win = w_ih + (size_t)(2 * Hh + j) * Hh;
        const float* whr = w_hh + (size_t)j * Hh;
        const float* whz = w_hh + (size_t)(Hh + j) * Hh;
        const float* whn = w_hh + (size_t)(2 * Hh + j) * Hh;
        for (int k = 0; k < Hh; ++k) {
            const float xv = xs[k], hv = hs[k];
            ir += wir[k] * xv; iz += wiz[k] * xv; inn += win[k] * xv;
            hr += whr[k] * hv; hz += whz[k] * hv; hnn += whn[k] * hv;
        }
        const float r = 1.f / (1.f + __expf(-(ir + hr)));
        const float z = 1.f / (1.f + __expf(-(iz + hz)));
        const float n = tanhf(inn + r * hnn);
        const float hv = (1.f - z) * n + z * hs[j];
        hn[j] = hv;
        h_new[row * Hh + j] = hv;
        h_new_b[(size_t)row * HPAD + j] = (__bf16)hv;
    }
    // zero the bf16 K-pad (16 elements)
    for (int j = Hh + tid; j < HPAD; j += 256)
        h_new_b[(size_t)row * HPAD + j] = (__bf16)0.0f;
    __syncthreads();

    // ---- masked dot-attention over encoder outputs
    const int len = enc_len[b];
    const float* eo = enc_out + (size_t)b * Ll * Hh;
    for (int l = tid; l < Ll; l += 256) {
        float d = 0.f;
        const float* er = eo + (size_t)l * Hh;
        for (int k = 0; k < Hh; ++k) d += hn[k] * er[k];
        sc[l] = (l < len) ? d : NEGINF;
    }
    __syncthreads();

    float mx = -INFINITY;
    for (int l = tid; l < Ll; l += 256) mx = fmaxf(mx, sc[l]);
    mx = block_reduce(mx, red, 1);
    float sum = 0.f;
    for (int l = tid; l < Ll; l += 256) sum += __expf(sc[l] - mx);
    sum = block_reduce(sum, red, 0);
    const float inv = 1.f / sum;
    for (int l = tid; l < Ll; l += 256) {
        const float p = __expf(sc[l] - mx) * inv;
        sc[l] = p;
        prob_g[row * Ll + l] = p;
    }
    __syncthreads();

    // ---- context = prob @ enc_out
    for (int j = tid; j < Hh; j += 256) {
        float c = 0.f;
        for (int l = 0; l < Ll; ++l) c += sc[l] * eo[(size_t)l * Hh + j];
        cx[j] = c;
    }
    __syncthreads();

    // ---- pointer switch = sigmoid(w_ratio . [h,ctx,x] + b)
    float part = 0.f;
    for (int j = tid; j < H3; j += 256) {
        const float v = (j < Hh) ? hn[j] : (j < 2 * Hh ? cx[j - Hh] : xs[j - 2 * Hh]);
        part += w_ratio_w[j] * v;
    }
    part = block_reduce(part, red, 0);
    const float sw = 1.f / (1.f + __expf(-(part + w_ratio_b[0])));
    if (tid == 0) sw_g[row] = sw;

    // ---- gate outputs from step-0 context
    if (t == 0) {
        for (int g = 0; g < NGg; ++g) {
            float pg = 0.f;
            for (int j = tid; j < Hh; j += 256) pg += w_gate_w[g * Hh + j] * cx[j];
            pg = block_reduce(pg, red, 0);
            if (tid == 0) gate_out[row * NGg + g] = pg + w_gate_b[g];
        }
    }
}

// ---------------------------------------------------------------------------
// WMMA fragment helpers (ISA 7.12.2 16-bit layouts, wave32)
// A (and mirrored B) fragment per lane = two runs of 8 consecutive K values:
//   K = kb + hiK + [0..8)  and  K = kb + 16 + hiK + [0..8),  hiK = 8*(lane>=16)

__device__ __forceinline__ v16bf combine16(v8bf lo, v8bf hi) {
    v16bf a;
    #pragma unroll
    for (int i = 0; i < 8; ++i) { a[i] = lo[i]; a[8 + i] = hi[i]; }
    return a;
}

__device__ __forceinline__ v16bf load_a_frag(const __bf16* As_row, int kb, int hiK) {
    const v8bf lo = *(const v8bf*)(As_row + kb + hiK);        // ds_load_b128
    const v8bf hi = *(const v8bf*)(As_row + kb + 16 + hiK);   // ds_load_b128
    return combine16(lo, hi);
}

__device__ __forceinline__ v16bf load_b_frag(const __bf16* brow, int kb, int hiK) {
    const v8bf lo = *(const v8bf*)(brow + kb + hiK);          // global_load_b128
    const v8bf hi = *(const v8bf*)(brow + kb + 16 + hiK);     // global_load_b128
    return combine16(lo, hi);
}

// ---------------------------------------------------------------------------
// Kernel 2: vocab logits via WMMA bf16:  logits[480,18000] = h[480,400] @ emb^T
// Block: 256 thr (8 waves). Block tile = 16 rows x 512 cols; wave tile = 16x64.
// Both operands pre-converted bf16, K zero-padded to 416 -> 13 uniform chunks.
// Software-pipelined: fragments for chunk kk+1 are fetched before the 4 WMMAs
// of chunk kk issue, so WMMAs never wait on their own operand loads.
// Writes raw logits directly into d_out slot [s,b,t,:].
__global__ void __launch_bounds__(256) vocab_logits_wmma(
    const __bf16* __restrict__ hb,     // [SB,HPAD] bf16
    const __bf16* __restrict__ embb,   // [V,HPAD]  bf16 (L2-resident, 15 MB)
    float*        __restrict__ out,    // [S,B,T,V]
    int t)
{
    __shared__ __bf16 As[16][HPAD];

    const int tid = threadIdx.x;
    const int m0  = blockIdx.y * 16;
    const int n0  = blockIdx.x * 512;

    // stage A tile: pure bf16 copy, 16B chunks
    for (int idx = tid; idx < 16 * HCH; idx += 256) {
        const int r  = idx / HCH;
        const int cb = (idx % HCH) * 8;
        *(v8bf*)&As[r][cb] = *(const v8bf*)(hb + (size_t)(m0 + r) * HPAD + cb);
    }
    __syncthreads();

    const int wave  = tid >> 5;        // wave32
    const int lane  = tid & 31;
    const int laneN = lane & 15;
    const int hiK   = (lane >> 4) * 8;
    const int nbase = n0 + wave * 64;

    // column ids of the 4 wave tiles; clamp load pointers (store is masked)
    const int n0c = nbase + laneN;
    const int n1c = nbase + 16 + laneN;
    const int n2c = nbase + 32 + laneN;
    const int n3c = nbase + 48 + laneN;
    const __bf16* brow0 = embb + (size_t)(n0c < Vv ? n0c : Vv - 1) * HPAD;
    const __bf16* brow1 = embb + (size_t)(n1c < Vv ? n1c : Vv - 1) * HPAD;
    const __bf16* brow2 = embb + (size_t)(n2c < Vv ? n2c : Vv - 1) * HPAD;
    const __bf16* brow3 = embb + (size_t)(n3c < Vv ? n3c : Vv - 1) * HPAD;

    const __bf16* As_row = &As[laneN][0];

    v8f acc0, acc1, acc2, acc3;
    #pragma unroll
    for (int r = 0; r < 8; ++r) { acc0[r] = 0.f; acc1[r] = 0.f; acc2[r] = 0.f; acc3[r] = 0.f; }

    // prologue: fragments for chunk 0
    v16bf a  = load_a_frag(As_row, 0, hiK);
    v16bf b0 = load_b_frag(brow0, 0, hiK);
    v16bf b1 = load_b_frag(brow1, 0, hiK);
    v16bf b2 = load_b_frag(brow2, 0, hiK);
    v16bf b3 = load_b_frag(brow3, 0, hiK);

    #pragma unroll
    for (int kk = 0; kk < 13; ++kk) {
        // consume current fragments
        const v16bf ca = a, c0 = b0, c1 = b1, c2 = b2, c3 = b3;
        // prefetch next chunk's fragments (issued before the WMMAs consume ca/c*)
        if (kk < 12) {
            const int kb = (kk + 1) * 32;
            a  = load_a_frag(As_row, kb, hiK);
            b0 = load_b_frag(brow0, kb, hiK);
            b1 = load_b_frag(brow1, kb, hiK);
            b2 = load_b_frag(brow2, kb, hiK);
            b3 = load_b_frag(brow3, kb, hiK);
        }
        acc0 = __builtin_amdgcn_wmma_f32_16x16x32_bf16(false, ca, false, c0, (short)0, acc0, false, false);
        acc1 = __builtin_amdgcn_wmma_f32_16x16x32_bf16(false, ca, false, c1, (short)0, acc1, false, false);
        acc2 = __builtin_amdgcn_wmma_f32_16x16x32_bf16(false, ca, false, c2, (short)0, acc2, false, false);
        acc3 = __builtin_amdgcn_wmma_f32_16x16x32_bf16(false, ca, false, c3, (short)0, acc3, false, false);
    }

    // C/D layout: lanes 0-15 -> M = r, lanes 16-31 -> M = r + 8; N = lane&15
    const int rowHalf = (lane < 16) ? 0 : 8;
    const int cols[4] = { n0c, n1c, n2c, n3c };
    v8f accs[4] = { acc0, acc1, acc2, acc3 };
    #pragma unroll
    for (int nt = 0; nt < 4; ++nt) {
        const int col = cols[nt];
        if (col < Vv) {
            #pragma unroll
            for (int r = 0; r < 8; ++r) {
                const int row = m0 + rowHalf + r;     // row = s*B+b
                out[(size_t)(row * Tt + t) * Vv + col] = accs[nt][r];
            }
        }
    }
}

// ---------------------------------------------------------------------------
// Kernel 3: in-place softmax over V, scale by switch, scatter-add pointer mass
// one block (256 thr) per row
__global__ void __launch_bounds__(256) softmax_scatter_kernel(
    float*       __restrict__ out,     // [S,B,T,V] (holds raw logits for step t)
    const float* __restrict__ prob_g,  // [SB,L]
    const float* __restrict__ sw_g,    // [SB]
    const int*   __restrict__ story,   // [B,L]
    int t)
{
    __shared__ float red[256];
    const int row = blockIdx.x;
    const int tid = threadIdx.x;
    const int b   = row % Bb;
    float* lg = out + (size_t)(row * Tt + t) * Vv;

    float mx = -INFINITY;
    for (int v = tid; v < Vv; v += 256) mx = fmaxf(mx, lg[v]);
    mx = block_reduce(mx, red, 1);

    float sum = 0.f;
    for (int v = tid; v < Vv; v += 256) sum += __expf(lg[v] - mx);
    sum = block_reduce(sum, red, 0);

    const float sw    = sw_g[row];
    const float scale = sw / sum;                       // switch * p_vocab
    for (int v = tid; v < Vv; v += 256) lg[v] = __expf(lg[v] - mx) * scale;
    __threadfence();
    __syncthreads();

    const float om = 1.0f - sw;                         // (1-switch) * p_ctx
    for (int l = tid; l < Ll; l += 256) {
        const float p  = prob_g[row * Ll + l];
        const int  tok = story[b * Ll + l];
        atomicAdd(&lg[tok], om * p);
    }
}

// ---------------------------------------------------------------------------
extern "C" void kernel_launch(void* const* d_in, const int* in_sizes, int n_in,
                              void* d_out, int out_size, void* d_ws, size_t ws_size,
                              hipStream_t stream) {
    (void)in_sizes; (void)n_in; (void)out_size; (void)ws_size;

    const float* enc_hidden = (const float*)d_in[0];
    const float* enc_out    = (const float*)d_in[1];
    const int*   enc_len    = (const int*)  d_in[2];
    const int*   story      = (const int*)  d_in[3];
    const int*   target     = (const int*)  d_in[4];
    const int*   domain_idx = (const int*)  d_in[5];
    const int*   slot_idx   = (const int*)  d_in[6];
    const float* emb        = (const float*)d_in[7];
    const float* slot_tab   = (const float*)d_in[8];
    const float* w_ih       = (const float*)d_in[9];
    const float* w_hh       = (const float*)d_in[10];
    const float* b_ih       = (const float*)d_in[11];
    const float* b_hh       = (const float*)d_in[12];
    const float* w_ratio_w  = (const float*)d_in[13];
    const float* w_ratio_b  = (const float*)d_in[14];
    const float* w_gate_w   = (const float*)d_in[15];
    const float* w_gate_b   = (const float*)d_in[16];

    float* out      = (float*)d_out;
    float* gate_out = out + (size_t)SBr * Tt * Vv;  // outputs concatenated flat

    // workspace layout (~17.7 MB)
    char* wsp = (char*)d_ws;
    __bf16* embb = (__bf16*)wsp;  wsp += (size_t)Vv  * HPAD * sizeof(__bf16);
    __bf16* hbf  = (__bf16*)wsp;  wsp += (size_t)SBr * HPAD * sizeof(__bf16);
    float*  hA   = (float*)wsp;   wsp += (size_t)SBr * Hh * sizeof(float);
    float*  hB   = (float*)wsp;   wsp += (size_t)SBr * Hh * sizeof(float);
    float*  prob = (float*)wsp;   wsp += (size_t)SBr * Ll * sizeof(float);
    float*  sw   = (float*)wsp;

    // one-shot: emb f32 -> bf16 padded (runs every call; deterministic)
    emb_to_bf16_kernel<<<(Vv * HCH + 255) / 256, 256, 0, stream>>>(emb, embb);

    const dim3 g2((Vv + 511) / 512, SBr / 16);   // 36 x 30 blocks

    for (int t = 0; t < Tt; ++t) {
        float* hcur  = (t & 1) ? hB : hA;
        float* hprev = (t & 1) ? hA : hB;  // unused at t==0 (kernel reads enc_hidden)

        gru_attn_kernel<<<SBr, 256, 0, stream>>>(
            enc_hidden, enc_out, enc_len, target, domain_idx, slot_idx,
            emb, slot_tab, w_ih, w_hh, b_ih, b_hh,
            w_ratio_w, w_ratio_b, w_gate_w, w_gate_b,
            hprev, hcur, hbf, prob, sw, gate_out, t);

        vocab_logits_wmma<<<g2, 256, 0, stream>>>(hbf, embb, out, t);

        softmax_scatter_kernel<<<SBr, 256, 0, stream>>>(out, prob, sw, story, t);
    }
}
